// DisplacementNet_63943473103528
// MI455X (gfx1250) — compile-verified, matrix operand order checked by reading, not run on hardware
//
#include <hip/hip_runtime.h>
#include <hip/hip_bf16.h>
#include <math.h>

#define NPTS 8192
#define KNN  12
#define COND 64
#define NUMF 8
#define GW   192
#define GL   4
#define MIXD 390
#define KPAD 416           // 13 * 32
#define KT   13            // K tiles of 32
#define NT   12            // 192 / 16 col tiles
#define CTW  4             // col tiles per wave (16x64 strip)
#define LN_EPS 1e-3f

typedef __attribute__((ext_vector_type(16))) _Float16 v16h;
typedef __attribute__((ext_vector_type(8)))  _Float16 v8h;
typedef __attribute__((ext_vector_type(8)))  float    v8f;

// ---------------------------------------------------------------------------
// Fourier features: feat[i][0:8]=sin(xB), [8:16]=cos(xB), [16:19]=x, [19]=0
// B_fourier shape (3, 8) row-major.
// ---------------------------------------------------------------------------
__global__ void feat_kernel(const float* __restrict__ x,
                            const float* __restrict__ Bf,
                            float* __restrict__ feat) {
    int i = blockIdx.x * blockDim.x + threadIdx.x;
    if (i >= NPTS) return;
    float x0 = x[i * 3 + 0], x1 = x[i * 3 + 1], x2 = x[i * 3 + 2];
#pragma unroll
    for (int m = 0; m < NUMF; ++m) {
        float xb = x0 * Bf[0 * NUMF + m] + x1 * Bf[1 * NUMF + m] + x2 * Bf[2 * NUMF + m];
        feat[i * 20 + m]        = sinf(xb);
        feat[i * 20 + NUMF + m] = cosf(xb);
    }
    feat[i * 20 + 16] = x0;
    feat[i * 20 + 17] = x1;
    feat[i * 20 + 18] = x2;
    feat[i * 20 + 19] = 0.0f;
}

// ---------------------------------------------------------------------------
// zproj[c] = bp[c] + sum_k z[k] * Wp[19+k][c]   (row-invariant part of proj)
// ---------------------------------------------------------------------------
__global__ void zproj_kernel(const float* __restrict__ z,
                             const float* __restrict__ Wp,
                             const float* __restrict__ bp,
                             float* __restrict__ zp) {
    int c = blockIdx.x * blockDim.x + threadIdx.x;
    if (c >= GW) return;
    float acc = bp[c];
#pragma unroll 8
    for (int k = 0; k < COND; ++k) acc += z[k] * Wp[(19 + k) * GW + c];
    zp[c] = acc;
}

// ---------------------------------------------------------------------------
// hcur[i][c] = zproj[c] + sum_{k<19} feat[i][k] * Wp[k][c]
// ---------------------------------------------------------------------------
__global__ void proj_kernel(const float* __restrict__ feat,
                            const float* __restrict__ Wp,
                            const float* __restrict__ zp,
                            float* __restrict__ hcur) {
    int idx = blockIdx.x * blockDim.x + threadIdx.x;
    if (idx >= NPTS * GW) return;
    int i = idx / GW, c = idx % GW;
    float acc = zp[c];
#pragma unroll
    for (int k = 0; k < 19; ++k) acc += feat[i * 20 + k] * Wp[k * GW + c];
    hcur[idx] = acc;
}

// ---------------------------------------------------------------------------
// Brute-force kNN: LDS-tiled distance scan, register-resident top-12
// (compare-swap chain with static indices -> stays in VGPRs).
// ---------------------------------------------------------------------------
__global__ __launch_bounds__(256) void knn_kernel(const float* __restrict__ x,
                                                  int* __restrict__ knn) {
    __shared__ float sx[256], sy[256], sz[256];
    int tid = threadIdx.x;
    int i = blockIdx.x * 256 + tid;
    float xi0 = x[i * 3 + 0], xi1 = x[i * 3 + 1], xi2 = x[i * 3 + 2];

    float bd[KNN];
    int   bi[KNN];
#pragma unroll
    for (int q = 0; q < KNN; ++q) { bd[q] = 3.0e38f; bi[q] = 0; }

    for (int t = 0; t < NPTS; t += 256) {
        int j = t + tid;
        sx[tid] = x[j * 3 + 0];
        sy[tid] = x[j * 3 + 1];
        sz[tid] = x[j * 3 + 2];
        __syncthreads();
        for (int jj = 0; jj < 256; ++jj) {
            int jg = t + jj;
            float d0 = xi0 - sx[jj];
            float d1 = xi1 - sy[jj];
            float d2 = xi2 - sz[jj];
            float d = d0 * d0 + d1 * d1 + d2 * d2;
            if (jg != i && d < bd[KNN - 1]) {
                float cd = d; int ci = jg;
#pragma unroll
                for (int q = 0; q < KNN; ++q) {
                    if (cd < bd[q]) {
                        float td = bd[q]; bd[q] = cd; cd = td;
                        int   ti = bi[q]; bi[q] = ci; ci = ti;
                    }
                }
            }
        }
        __syncthreads();
    }
#pragma unroll
    for (int q = 0; q < KNN; ++q) knn[i * KNN + q] = bi[q];
}

// ---------------------------------------------------------------------------
// Pack Wg[layer] (390x192 f32) into WMMA B-fragment order, f16, K padded 416.
// Layout: Bpack[((layer*13 + kt)*12 + nt)*32 + lane][v], v=0..15, 32B/lane.
// K index inside a 32-deep tile for lane/v: (v>>3)*16 + (lane>>4)*8 + (v&7).
// ---------------------------------------------------------------------------
__global__ __launch_bounds__(32) void pack_wg_kernel(const float* __restrict__ Wg,
                                                     _Float16* __restrict__ Bpack) {
    int kt = blockIdx.x, nt = blockIdx.y, layer = blockIdx.z;
    int lane = threadIdx.x;
    int col  = nt * 16 + (lane & 15);
    int half = lane >> 4;
    _Float16* dst = Bpack + ((((size_t)layer * KT + kt) * NT + nt) * 32 + lane) * 16;
#pragma unroll
    for (int v = 0; v < 16; ++v) {
        int k = kt * 32 + (v >> 3) * 16 + half * 8 + (v & 7);
        float val = (k < MIXD) ? Wg[((size_t)layer * MIXD + k) * GW + col] : 0.0f;
        dst[v] = (_Float16)val;
    }
}

// ---------------------------------------------------------------------------
// Gather + aggregate -> mix row (f16, padded to 416):
// [hcur(192) | mean_nbr_hcur(192) | rel_mean(3) | rel_std(3) | zeros(26)]
// One block (256 threads) per point.
// ---------------------------------------------------------------------------
__global__ __launch_bounds__(256) void gather_mix_kernel(const float* __restrict__ x,
                                                         const float* __restrict__ hcur,
                                                         const int* __restrict__ knn,
                                                         _Float16* __restrict__ mixh) {
    int i = blockIdx.x;
    int tid = threadIdx.x;
    __shared__ int nbr[KNN];
    if (tid < KNN) nbr[tid] = knn[i * KNN + tid];
    __syncthreads();
    _Float16* mrow = mixh + (size_t)i * KPAD;

    if (tid < GW) {
        float hv = hcur[(size_t)i * GW + tid];
        float s = 0.0f;
#pragma unroll
        for (int k = 0; k < KNN; ++k) s += hcur[(size_t)nbr[k] * GW + tid];
        mrow[tid]      = (_Float16)hv;
        mrow[GW + tid] = (_Float16)(s * (1.0f / (float)KNN));
    } else if (tid < GW + 6) {
        int d = (tid - GW) % 3;
        bool isStd = (tid - GW) >= 3;
        float xc = x[i * 3 + d];
        float vals[KNN];
        float m = 0.0f;
#pragma unroll
        for (int k = 0; k < KNN; ++k) { vals[k] = x[nbr[k] * 3 + d] - xc; m += vals[k]; }
        m *= (1.0f / (float)KNN);
        if (!isStd) {
            mrow[2 * GW + d] = (_Float16)m;
        } else {
            float v = 0.0f;
#pragma unroll
            for (int k = 0; k < KNN; ++k) { float t = vals[k] - m; v += t * t; }
            mrow[2 * GW + 3 + d] = (_Float16)sqrtf(v * (1.0f / (float)KNN));
        }
    } else if (tid < GW + 6 + (KPAD - MIXD)) {   // zero pad cols 390..415
        mrow[MIXD + (tid - GW - 6)] = (_Float16)0.0f;
    }
}

// ---------------------------------------------------------------------------
// WMMA GEMM + bias + SiLU, 16x64 output strip per wave:
//   hcur(8192x192,f32) = silu( mix(8192x416,f16) @ Bpack[layer] + bg[layer] )
// Per K-step: 1 A fragment (2 x b128), 4 B fragments (8 x b128), 4 independent
// v_wmma_f32_16x16x32_f16 into 4 accumulators -> 2.5 loads per WMMA and no
// WMMA->WMMA register dependency inside a K-step.
// ---------------------------------------------------------------------------
__global__ __launch_bounds__(32) void gemm_silu_wmma_kernel(const _Float16* __restrict__ mixh,
                                                            const _Float16* __restrict__ Bpack,
                                                            const float* __restrict__ bg,
                                                            float* __restrict__ hcur,
                                                            int layer) {
    int rowTile = blockIdx.x;          // 0..511
    int ctg     = blockIdx.y;          // 0..2  (group of 4 col tiles)
    int lane = threadIdx.x;            // 0..31
    int half = lane >> 4;
    int r16  = lane & 15;

    const _Float16* arow  = mixh + (size_t)(rowTile * 16 + r16) * KPAD;
    const _Float16* bbase = Bpack +
        (((size_t)layer * KT * NT + (size_t)ctg * CTW) * 32 + lane) * 16;
    const size_t bstride_kt = (size_t)NT * 32 * 16;   // advance one K-tile
    const size_t bstride_nt = (size_t)32 * 16;        // advance one col tile

    v8f acc[CTW];
#pragma unroll
    for (int t = 0; t < CTW; ++t) acc[t] = (v8f){};

#pragma unroll
    for (int kt = 0; kt < KT; ++kt) {
        union { v16h v; v8h h[2]; } A;
        A.h[0] = *(const v8h*)(arow + kt * 32 + half * 8);
        A.h[1] = *(const v8h*)(arow + kt * 32 + 16 + half * 8);
        const _Float16* bkt = bbase + (size_t)kt * bstride_kt;
        v16h B0 = *(const v16h*)(bkt + 0 * bstride_nt);
        v16h B1 = *(const v16h*)(bkt + 1 * bstride_nt);
        v16h B2 = *(const v16h*)(bkt + 2 * bstride_nt);
        v16h B3 = *(const v16h*)(bkt + 3 * bstride_nt);
        if (kt + 1 < KT)
            __builtin_prefetch(bbase + (size_t)(kt + 1) * bstride_kt, 0, 1);
        acc[0] = __builtin_amdgcn_wmma_f32_16x16x32_f16(false, A.v, false, B0,
                                                        (short)0, acc[0], false, false);
        acc[1] = __builtin_amdgcn_wmma_f32_16x16x32_f16(false, A.v, false, B1,
                                                        (short)0, acc[1], false, false);
        acc[2] = __builtin_amdgcn_wmma_f32_16x16x32_f16(false, A.v, false, B2,
                                                        (short)0, acc[2], false, false);
        acc[3] = __builtin_amdgcn_wmma_f32_16x16x32_f16(false, A.v, false, B3,
                                                        (short)0, acc[3], false, false);
    }

#pragma unroll
    for (int t = 0; t < CTW; ++t) {
        int col = (ctg * CTW + t) * 16 + r16;
        float bias = bg[layer * GW + col];
#pragma unroll
        for (int r = 0; r < 8; ++r) {
            int row = rowTile * 16 + r + 8 * half;
            float v = acc[t][r] + bias;
            float s = v / (1.0f + __expf(-v));     // SiLU
            hcur[(size_t)row * GW + col] = s;
        }
    }
}

// ---------------------------------------------------------------------------
// LayerNorm (eps=1e-3) + output projection 192 -> 3. One thread per point.
// ---------------------------------------------------------------------------
__global__ void final_kernel(const float* __restrict__ hcur,
                             const float* __restrict__ gamma,
                             const float* __restrict__ beta,
                             const float* __restrict__ Wo,
                             const float* __restrict__ bo,
                             float* __restrict__ out) {
    int i = blockIdx.x * blockDim.x + threadIdx.x;
    if (i >= NPTS) return;
    const float* hr = hcur + (size_t)i * GW;
    float mu = 0.0f;
#pragma unroll 8
    for (int c = 0; c < GW; ++c) mu += hr[c];
    mu *= (1.0f / (float)GW);
    float var = 0.0f;
#pragma unroll 8
    for (int c = 0; c < GW; ++c) { float t = hr[c] - mu; var += t * t; }
    var *= (1.0f / (float)GW);
    float inv = rsqrtf(var + LN_EPS);
    float o0 = bo[0], o1 = bo[1], o2 = bo[2];
#pragma unroll 4
    for (int c = 0; c < GW; ++c) {
        float hn = (hr[c] - mu) * inv * gamma[c] + beta[c];
        o0 += hn * Wo[c * 3 + 0];
        o1 += hn * Wo[c * 3 + 1];
        o2 += hn * Wo[c * 3 + 2];
    }
    out[i * 3 + 0] = o0;
    out[i * 3 + 1] = o1;
    out[i * 3 + 2] = o2;
}

// ---------------------------------------------------------------------------
// Launch
// ---------------------------------------------------------------------------
extern "C" void kernel_launch(void* const* d_in, const int* in_sizes, int n_in,
                              void* d_out, int out_size, void* d_ws, size_t ws_size,
                              hipStream_t stream) {
    (void)in_sizes; (void)n_in; (void)out_size; (void)ws_size;

    const float* x     = (const float*)d_in[0];   // (8192,3)
    const float* z     = (const float*)d_in[1];   // (64,)
    const float* Bf    = (const float*)d_in[2];   // (3,8)
    const float* Wp    = (const float*)d_in[3];   // (83,192)
    const float* bp    = (const float*)d_in[4];   // (192,)
    const float* Wg    = (const float*)d_in[5];   // (4,390,192)
    const float* bg    = (const float*)d_in[6];   // (4,192)
    const float* gamma = (const float*)d_in[7];   // (192,)
    const float* beta  = (const float*)d_in[8];   // (192,)
    const float* Wo    = (const float*)d_in[9];   // (192,3)
    const float* bo    = (const float*)d_in[10];  // (3,)
    float* out = (float*)d_out;                   // (8192,3) f32

    // Workspace layout (all 256-byte aligned)
    char* ws = (char*)d_ws;
    size_t off = 0;
    auto carve = [&](size_t bytes) -> char* {
        char* p = ws + off;
        off += (bytes + 255) & ~(size_t)255;
        return p;
    };
    int*      knn   = (int*)     carve((size_t)NPTS * KNN * sizeof(int));        // 384 KB
    float*    feat  = (float*)   carve((size_t)NPTS * 20 * sizeof(float));       // 640 KB
    float*    hcur  = (float*)   carve((size_t)NPTS * GW * sizeof(float));       // 6 MB
    _Float16* mixh  = (_Float16*)carve((size_t)NPTS * KPAD * sizeof(_Float16));  // 6.5 MB
    _Float16* Bpack = (_Float16*)carve((size_t)GL * KT * NT * 32 * 16 * 2);      // 624 KB
    float*    zp    = (float*)   carve((size_t)GW * sizeof(float));              // tiny

    // Weight prep (packed WMMA B fragments) + row-invariant z projection
    hipLaunchKernelGGL(pack_wg_kernel, dim3(KT, NT, GL), dim3(32), 0, stream, Wg, Bpack);
    hipLaunchKernelGGL(zproj_kernel, dim3(1), dim3(GW), 0, stream, z, Wp, bp, zp);

    // Features, kNN graph, input projection
    hipLaunchKernelGGL(feat_kernel, dim3(NPTS / 256), dim3(256), 0, stream, x, Bf, feat);
    hipLaunchKernelGGL(knn_kernel, dim3(NPTS / 256), dim3(256), 0, stream, x, knn);
    hipLaunchKernelGGL(proj_kernel, dim3((NPTS * GW) / 256), dim3(256), 0, stream,
                       feat, Wp, zp, hcur);

    // 4 graph layers: gather/aggregate -> WMMA GEMM + SiLU (in-place hcur update)
    for (int l = 0; l < GL; ++l) {
        hipLaunchKernelGGL(gather_mix_kernel, dim3(NPTS), dim3(256), 0, stream,
                           x, hcur, knn, mixh);
        hipLaunchKernelGGL(gemm_silu_wmma_kernel, dim3(NPTS / 16, NT / CTW), dim3(32),
                           0, stream, mixh, Bpack, bg, hcur, l);
    }

    // LayerNorm + output head
    hipLaunchKernelGGL(final_kernel, dim3(NPTS / 256), dim3(256), 0, stream,
                       hcur, gamma, beta, Wo, bo, out);
}
